// RAGModule_29231547417032
// MI455X (gfx1250) — matrix operand order repaired and checked by reading, not compiled, not run on hardware
//
#include <hip/hip_runtime.h>
#include <hip/hip_bf16.h>

typedef __attribute__((ext_vector_type(16))) __bf16 v16bf;
typedef __attribute__((ext_vector_type(8)))  __bf16 v8bf;
typedef __attribute__((ext_vector_type(8)))  float  v8f;

#define B_ROWS 256
#define DDIM   64
#define TOPK   3
#define TILE_BF_STRIDE 72   // bf16 elems: 144B row stride -> conflict-free, 16B aligned
#define NEG_INF (-3.4e38f)

// Branchless sorted top-3 insert (compiles to v_cmp + v_cndmask chains).
__device__ __forceinline__ void top3_insert_sel(float v, int i,
    float& v0, int& i0, float& v1, int& i1, float& v2, int& i2) {
  const bool g0 = v > v0, g1 = v > v1, g2 = v > v2;
  const float nv0 = g0 ? v  : v0;
  const int   ni0 = g0 ? i  : i0;
  const float nv1 = g0 ? v0 : (g1 ? v : v1);
  const int   ni1 = g0 ? i0 : (g1 ? i : i1);
  const float nv2 = g1 ? v1 : (g2 ? v : v2);
  const int   ni2 = g1 ? i1 : (g2 ? i : i2);
  v0 = nv0; i0 = ni0; v1 = nv1; i1 = ni1; v2 = nv2; i2 = ni2;
}

// ---------------------------------------------------------------------------
// K1: q_unit = emb / (norm>0 ? norm : 1) -> bf16, row-major [256][64]
// ---------------------------------------------------------------------------
__global__ __launch_bounds__(64) void qnorm_bf16_kernel(
    const float* __restrict__ emb, __bf16* __restrict__ qbf) {
  __shared__ float red[64];
  const int r = blockIdx.x, d = threadIdx.x;
  const float x = emb[r * DDIM + d];
  red[d] = x * x;
  __syncthreads();
  for (int s = 32; s > 0; s >>= 1) { if (d < s) red[d] += red[d + s]; __syncthreads(); }
  const float n2 = red[0];
  const float sc = (n2 > 0.f) ? rsqrtf(n2) : 1.f;
  qbf[r * DDIM + d] = (__bf16)(x * sc);
}

// ---------------------------------------------------------------------------
// K2: streaming cosine sims via bf16 WMMA + screened running top-3.
// Grid: nwg x 256 threads (8 waves). Each wave owns two 16-row query tiles
// (persistent A fragments); per 16-column KB tile: bf16-convert at load into
// LDS, shuffle-reduced row norms, 4x v_wmma_f32_16x16x32_bf16, then a cheap
// max-screen before the (rare) branchless top-3 inserts.
// ---------------------------------------------------------------------------
__global__ __launch_bounds__(256) void sims_topk_kernel(
    const float* __restrict__ kb, const __bf16* __restrict__ qbf,
    float* __restrict__ candV, int* __restrict__ candI, int nkb) {
  __shared__ __align__(16) __bf16 tileb[16 * TILE_BF_STRIDE];
  __shared__ float rns[16];

  const int tid  = threadIdx.x;
  const int lane = tid & 31;
  const int wv   = tid >> 5;
  const int half = lane >> 4;   // lane group 0/1
  const int l15  = lane & 15;
  const int tidr = tid >> 4;    // KB tile row this thread loads (0..15)
  const int tidq = tid & 15;    // 4-float quarter within the row

  // Persistent A fragments. A 16x32 bf16 layout per lane is two contiguous
  // 8-element K runs: base = (half?8:0)+32h, elems 0..7 = K base..base+7,
  // elems 8..15 = K base+16..base+23. Load as 2x16B.
  v16bf a[2][2];
#pragma unroll
  for (int j = 0; j < 2; ++j) {
    const int row = (wv * 2 + j) * 16 + l15;
#pragma unroll
    for (int h = 0; h < 2; ++h) {
      const __bf16* p = qbf + row * DDIM + (half ? 8 : 0) + 32 * h;
      const v8bf lo = *(const v8bf*)(p);
      const v8bf hi = *(const v8bf*)(p + 16);
      a[j][h] = __builtin_shufflevector(lo, hi, 0, 1, 2, 3, 4, 5, 6, 7,
                                        8, 9, 10, 11, 12, 13, 14, 15);
    }
  }

  // Running per-lane top-3 for (2 M-tiles) x (8 C rows) + screen thresholds
  float tv[2][8][3];
  int   ti[2][8][3];
  float thr[2] = {NEG_INF, NEG_INF};
#pragma unroll
  for (int j = 0; j < 2; ++j)
#pragma unroll
    for (int r = 0; r < 8; ++r)
#pragma unroll
      for (int k = 0; k < 3; ++k) { tv[j][r][k] = NEG_INF; ti[j][r][k] = 0; }

  const int ntiles = (nkb + 15) >> 4;

  // Register double buffer: preload first tile.
  int t = blockIdx.x;
  float4 cur = make_float4(0.f, 0.f, 0.f, 0.f);
  if (t < ntiles) {
    const int kr = (t << 4) + tidr;
    if (kr < nkb) cur = *(const float4*)(kb + (size_t)kr * DDIM + tidq * 4);
  }

  for (; t < ntiles; t += gridDim.x) {
    const int n0 = t << 4;
    __syncthreads();                      // previous iteration's readers done

    // Row sum-of-squares: half-wave shuffle reduction (16 loaders per row).
    float ss = cur.x * cur.x + cur.y * cur.y + cur.z * cur.z + cur.w * cur.w;
#pragma unroll
    for (int m = 1; m <= 8; m <<= 1) ss += __shfl_xor(ss, m, 32);
    if (tidq == 0) rns[tidr] = (ss > 0.f) ? rsqrtf(ss) : 1.f;

    // bf16-pack in registers, single 8B LDS store.
    union { __bf16 h[4]; uint2 u; } pk;
    pk.h[0] = (__bf16)cur.x; pk.h[1] = (__bf16)cur.y;
    pk.h[2] = (__bf16)cur.z; pk.h[3] = (__bf16)cur.w;
    *(uint2*)(&tileb[tidr * TILE_BF_STRIDE + tidq * 4]) = pk.u;

    // Issue next tile's global load now (retires under the WMMA phase).
    const int tn = t + gridDim.x;
    float4 nxt = make_float4(0.f, 0.f, 0.f, 0.f);
    if (tn < ntiles) {
      const int kr = (tn << 4) + tidr;
      if (kr < nkb) nxt = *(const float4*)(kb + (size_t)kr * DDIM + tidq * 4);
    }
    // L2 prefetch two tiles ahead (-> global_prefetch_b8).
    const int tp = t + 2 * gridDim.x;
    if (tp < ntiles) {
      const int kr = (tp << 4) + tidr;
      if (kr < nkb) __builtin_prefetch(kb + (size_t)kr * DDIM + tidq * 4, 0, 1);
    }
    __syncthreads();                      // tile + rns visible

    // B fragments straight from bf16 LDS: two 8-elem K runs (base, base+16), matching A.
    v16bf b[2];
#pragma unroll
    for (int h = 0; h < 2; ++h) {
      const __bf16* p = &tileb[l15 * TILE_BF_STRIDE + (half ? 8 : 0) + 32 * h];
      const v8bf lo = *(const v8bf*)(p);
      const v8bf hi = *(const v8bf*)(p + 16);
      b[h] = __builtin_shufflevector(lo, hi, 0, 1, 2, 3, 4, 5, 6, 7,
                                     8, 9, 10, 11, 12, 13, 14, 15);
    }

    const float rn  = rns[l15];
    const int   idx = n0 + l15;
    const bool  ok  = idx < nkb;

    v8f c[2];
#pragma unroll
    for (int j = 0; j < 2; ++j) {
      v8f z = {0.f, 0.f, 0.f, 0.f, 0.f, 0.f, 0.f, 0.f};
      z = __builtin_amdgcn_wmma_f32_16x16x32_bf16(false, a[j][0], false, b[0],
                                                  (short)0, z, false, false);
      z = __builtin_amdgcn_wmma_f32_16x16x32_bf16(false, a[j][1], false, b[1],
                                                  (short)0, z, false, false);
      c[j] = z;
    }

#pragma unroll
    for (int j = 0; j < 2; ++j) {
      float vs[8];
      float vmax = NEG_INF;
#pragma unroll
      for (int r = 0; r < 8; ++r) {
        vs[r] = c[j][r] * rn;             // fold 1/||k|| post-WMMA (f32)
        vmax = fmaxf(vmax, vs[r]);
      }
      // Screen: only touch the sorted triples if something can change.
      if (ok && vmax > thr[j]) {
#pragma unroll
        for (int r = 0; r < 8; ++r)
          top3_insert_sel(vs[r], idx,
                          tv[j][r][0], ti[j][r][0],
                          tv[j][r][1], ti[j][r][1],
                          tv[j][r][2], ti[j][r][2]);
        float m = tv[j][0][2];
#pragma unroll
        for (int r = 1; r < 8; ++r) m = fminf(m, tv[j][r][2]);
        thr[j] = m;
      }
    }

    cur = nxt;
  }

  // Merge the 16 column-slots of each row with a shuffle-xor tree (wave32).
#pragma unroll
  for (int j = 0; j < 2; ++j) {
#pragma unroll
    for (int r = 0; r < 8; ++r) {
      float v0 = tv[j][r][0], v1 = tv[j][r][1], v2 = tv[j][r][2];
      int   i0 = ti[j][r][0], i1 = ti[j][r][1], i2 = ti[j][r][2];
#pragma unroll
      for (int m = 1; m <= 8; m <<= 1) {
        const float u0 = __shfl_xor(v0, m, 32);
        const float u1 = __shfl_xor(v1, m, 32);
        const float u2 = __shfl_xor(v2, m, 32);
        const int   b0 = __shfl_xor(i0, m, 32);
        const int   b1 = __shfl_xor(i1, m, 32);
        const int   b2 = __shfl_xor(i2, m, 32);
        top3_insert_sel(u0, b0, v0, i0, v1, i1, v2, i2);
        top3_insert_sel(u1, b1, v0, i0, v1, i1, v2, i2);
        top3_insert_sel(u2, b2, v0, i0, v1, i1, v2, i2);
      }
      if (l15 == 0) {
        const int row  = (wv * 2 + j) * 16 + r + (half ? 8 : 0);
        const int base = (blockIdx.x * B_ROWS + row) * TOPK;
        candV[base + 0] = v0; candI[base + 0] = i0;
        candV[base + 1] = v1; candI[base + 1] = i1;
        candV[base + 2] = v2; candI[base + 2] = i2;
      }
    }
  }
}

// ---------------------------------------------------------------------------
// K3: per-row global top-3 from per-WG candidates; gather + mean -> context
// ---------------------------------------------------------------------------
__global__ __launch_bounds__(64) void reduce_ctx_kernel(
    const float* __restrict__ kb, const float* __restrict__ candV,
    const int* __restrict__ candI, float* __restrict__ ctx, int nwg) {
  __shared__ float lv[64 * 3];
  __shared__ int   li[64 * 3];
  __shared__ int   sel[3];
  const int q = blockIdx.x, t = threadIdx.x;
  float v0 = NEG_INF, v1 = NEG_INF, v2 = NEG_INF;
  int   i0 = 0, i1 = 0, i2 = 0;
  for (int wg = t; wg < nwg; wg += 64) {
    const int base = (wg * B_ROWS + q) * TOPK;
#pragma unroll
    for (int k = 0; k < 3; ++k)
      top3_insert_sel(candV[base + k], candI[base + k], v0, i0, v1, i1, v2, i2);
  }
  lv[t * 3 + 0] = v0; li[t * 3 + 0] = i0;
  lv[t * 3 + 1] = v1; li[t * 3 + 1] = i1;
  lv[t * 3 + 2] = v2; li[t * 3 + 2] = i2;
  __syncthreads();
  if (t == 0) {
    float m0 = NEG_INF, m1 = NEG_INF, m2 = NEG_INF; int j0 = 0, j1 = 0, j2 = 0;
    for (int s = 0; s < 64 * 3; ++s)
      top3_insert_sel(lv[s], li[s], m0, j0, m1, j1, m2, j2);
    sel[0] = j0; sel[1] = j1; sel[2] = j2;
  }
  __syncthreads();
  ctx[q * DDIM + t] = (kb[(size_t)sel[0] * DDIM + t] +
                       kb[(size_t)sel[1] * DDIM + t] +
                       kb[(size_t)sel[2] * DDIM + t]) * (1.f / 3.f);
}

// ---------------------------------------------------------------------------
// K4: MLP head. attn = exp(s - s) == 1, so q/k are dead:
// attended = (ctx @ wv.T + bv) @ out_w.T + out_b, then the ci/cls chain.
// ---------------------------------------------------------------------------
__global__ __launch_bounds__(64) void mlp_head_kernel(
    const float* __restrict__ emb, const float* __restrict__ ctx,
    const float* __restrict__ in_w, const float* __restrict__ in_b,
    const float* __restrict__ out_w, const float* __restrict__ out_b,
    const float* __restrict__ ci_w1, const float* __restrict__ ci_b1,
    const float* __restrict__ ci_w2, const float* __restrict__ ci_b2,
    const float* __restrict__ cls_w1, const float* __restrict__ cls_b1,
    const float* __restrict__ cls_w2, const float* __restrict__ cls_b2,
    float* __restrict__ pred) {
  __shared__ float se[64], sx[64], bufA[64], bufB[64], cbuf[128];
  const int q = blockIdx.x, t = threadIdx.x;
  se[t] = emb[q * DDIM + t];
  sx[t] = ctx[q * DDIM + t];
  __syncthreads();
  float acc = in_b[128 + t];
  for (int d = 0; d < 64; ++d) acc += sx[d] * in_w[(128 + t) * 64 + d];
  bufA[t] = acc;                                       // v
  __syncthreads();
  acc = out_b[t];
  for (int d = 0; d < 64; ++d) acc += bufA[d] * out_w[t * 64 + d];
  bufB[t] = acc;                                       // attended
  __syncthreads();
  acc = ci_b1[t];
  for (int d = 0; d < 64; ++d) acc += se[d]   * ci_w1[t * 128 + d];
  for (int d = 0; d < 64; ++d) acc += bufB[d] * ci_w1[t * 128 + 64 + d];
  bufA[t] = fmaxf(acc, 0.f);                           // h
  __syncthreads();
  acc = ci_b2[t];
  for (int d = 0; d < 64; ++d) acc += bufA[d] * ci_w2[t * 64 + d];
  bufB[t] = acc;                                       // enhanced
  __syncthreads();
#pragma unroll
  for (int k = 0; k < 2; ++k) {
    const int o = t + 64 * k;
    float a2 = cls_b1[o];
    for (int d = 0; d < 64; ++d) a2 += bufB[d] * cls_w1[o * 64 + d];
    cbuf[o] = fmaxf(a2, 0.f);                          // c
  }
  __syncthreads();
  if (t < 2) {
    float a2 = cls_b2[t];
    for (int d = 0; d < 128; ++d) a2 += cbuf[d] * cls_w2[t * 128 + d];
    pred[q * 2 + t] = a2;
  }
}

// ---------------------------------------------------------------------------
extern "C" void kernel_launch(void* const* d_in, const int* in_sizes, int n_in,
                              void* d_out, int out_size, void* d_ws, size_t ws_size,
                              hipStream_t stream) {
  const float* emb    = (const float*)d_in[0];
  const float* kb     = (const float*)d_in[1];
  const float* in_w   = (const float*)d_in[2];
  const float* in_b   = (const float*)d_in[3];
  const float* out_w  = (const float*)d_in[4];
  const float* out_b  = (const float*)d_in[5];
  const float* ci_w1  = (const float*)d_in[6];
  const float* ci_b1  = (const float*)d_in[7];
  const float* ci_w2  = (const float*)d_in[8];
  const float* ci_b2  = (const float*)d_in[9];
  const float* cls_w1 = (const float*)d_in[10];
  const float* cls_b1 = (const float*)d_in[11];
  const float* cls_w2 = (const float*)d_in[12];
  const float* cls_b2 = (const float*)d_in[13];
  float* out = (float*)d_out;                 // [256*2 preds][256*64 context]
  const int nkb = in_sizes[1] / DDIM;

  // Workspace layout: bf16 q_unit | candV | candI
  char* ws = (char*)d_ws;
  __bf16* qbf = (__bf16*)ws;
  const size_t qbytes = (size_t)B_ROWS * DDIM * sizeof(__bf16);
  const size_t avail  = (ws_size > qbytes) ? (ws_size - qbytes) : 0;
  int nwg = (int)(avail / ((size_t)B_ROWS * TOPK * 8));
  if (nwg > 512) nwg = 512;
  if (nwg < 1)   nwg = 1;
  float* candV = (float*)(ws + qbytes);
  int*   candI = (int*)(ws + qbytes + (size_t)nwg * B_ROWS * TOPK * sizeof(float));

  float* ctx_out  = out + B_ROWS * 2;
  float* pred_out = out;

  qnorm_bf16_kernel<<<B_ROWS, 64, 0, stream>>>(emb, qbf);
  sims_topk_kernel<<<nwg, 256, 0, stream>>>(kb, qbf, candV, candI, nkb);
  reduce_ctx_kernel<<<B_ROWS, 64, 0, stream>>>(kb, candV, candI, ctx_out, nwg);
  mlp_head_kernel<<<B_ROWS, 64, 0, stream>>>(emb, ctx_out, in_w, in_b, out_w, out_b,
                                             ci_w1, ci_b1, ci_w2, ci_b2,
                                             cls_w1, cls_b1, cls_w2, cls_b2, pred_out);
}
